// MaskedSelfAttention_48576080118520
// MI455X (gfx1250) — matrix operand order repaired and checked by reading, not compiled, not run on hardware
//
#include <hip/hip_runtime.h>

#define DIMV   1024
#define NHEADS 16
#define HDIM   64
#define BB     4
#define SS     2048
#define MTOT   (BB*SS)          // 8192 rows
#define EPSV   1e-6f

typedef __attribute__((ext_vector_type(16))) __bf16 v16bf;
typedef __attribute__((ext_vector_type(8)))  float  v8f;

union FragB16 { unsigned int u[8]; v16bf v; };

// f32 -> bf16 via native conversion (hardware v_cvt on gfx1250)
__device__ __forceinline__ unsigned short f2bf(float f) {
    union { __bf16 b; unsigned short u; } c;
    c.b = (__bf16)f;
    return c.u;
}
__device__ __forceinline__ unsigned pack2(unsigned short lo, unsigned short hi) {
    return (unsigned)lo | ((unsigned)hi << 16);
}
__device__ __forceinline__ unsigned f2bf2(float lo, float hi) {
    return pack2(f2bf(lo), f2bf(hi));
}

// A-matrix (16x32 bf16) per-lane K map: element pair j (0..7) covers K = {kA, kA+1}
// (contiguous: lanes 0-15 hold K 0..15, lanes 16-31 hold K 16..31, matching B)
__device__ __forceinline__ int kAmap(int j, int hi8) {
    return 2*hi8 + 2*j;
}

// Load one B-fragment half-column: 16 consecutive f32 along K, convert to bf16.
// B layout: lane holds column N=lane&15; lanes 0-15 hold K rows 0..15, lanes 16-31 K 16..31.
__device__ __forceinline__ FragB16 load_b_from_f32(const float* __restrict__ wk) {
    FragB16 b;
    const float4* p = (const float4*)wk;
#pragma unroll
    for (int q = 0; q < 4; ++q) {
        float4 f = p[q];
        b.u[2*q+0] = f2bf2(f.x, f.y);
        b.u[2*q+1] = f2bf2(f.z, f.w);
    }
    return b;
}

// Load one B-fragment half-column from bf16 memory where the 16 K elements are
// contiguous (32 bytes).
__device__ __forceinline__ FragB16 load_b_contig_bf16(const unsigned short* __restrict__ p) {
    FragB16 b;
    uint4 lo = ((const uint4*)p)[0];
    uint4 hi = ((const uint4*)p)[1];
    b.u[0]=lo.x; b.u[1]=lo.y; b.u[2]=lo.z; b.u[3]=lo.w;
    b.u[4]=hi.x; b.u[5]=hi.y; b.u[6]=hi.z; b.u[7]=hi.w;
    return b;
}

// ---------------------------------------------------------------------------
// Kernel 1: QKV GEMM (f32 in, bf16 WMMA, f32 acc) + bias + fused per-head
// RMS-norm epilogue. Block tile 32(M) x 64(N); N-block == one (t,h) pair so the
// whole Dh=64 norm row lives in the block's tile.
// Outputs: q̂*scale, k̂ as bf16 [b][h][s][dh];  v as bf16 TRANSPOSED [b][h][dh][s]
// so the PV B-fragments in attention are contiguous along the key axis.
// ---------------------------------------------------------------------------
__global__ __launch_bounds__(128)
void qkv_norm_kernel(const float* __restrict__ X,      // [MTOT, DIMV]
                     const float* __restrict__ W,      // [3*DIMV, DIMV] (e-major)
                     const float* __restrict__ bias,   // [3*DIMV]
                     const float* __restrict__ qw,     // [HDIM]
                     const float* __restrict__ kw,     // [HDIM]
                     unsigned short* __restrict__ Qn,  // [B,H,S,HDIM] bf16
                     unsigned short* __restrict__ Kn,  // [B,H,S,HDIM] bf16
                     unsigned short* __restrict__ Vt)  // [B,H,HDIM,S] bf16
{
    __shared__ unsigned short sA[32][32];   // bf16 A tile
    __shared__ float          sC[32][64];   // f32 C tile for norm epilogue
    __shared__ float          sRstd[32];

    const int tid    = threadIdx.x;
    const int wave   = tid >> 5;
    const int lane   = tid & 31;
    const int lane16 = lane & 15;
    const int hi8    = (lane >> 4) << 3;     // 0 or 8
    const int n0     = blockIdx.x * 64;      // head-aligned e-tile base
    const int m0     = blockIdx.y * 32;

    v8f c0 = {}; v8f c1 = {};

    for (int k0 = 0; k0 < DIMV; k0 += 32) {
        // stage A tile (f32 -> bf16, packed b32 LDS stores)
        {
            int row = tid >> 2;
            int cb  = (tid & 3) * 8;
            const float4* p = (const float4*)(X + (size_t)(m0 + row)*DIMV + k0 + cb);
            float4 f0 = p[0], f1 = p[1];
            unsigned* d = (unsigned*)&sA[row][cb];
            d[0] = f2bf2(f0.x, f0.y);
            d[1] = f2bf2(f0.z, f0.w);
            d[2] = f2bf2(f1.x, f1.y);
            d[3] = f2bf2(f1.z, f1.w);
            if (k0 + 32 < DIMV)
                __builtin_prefetch(X + (size_t)(m0 + row)*DIMV + k0 + 32 + cb, 0, 0);
        }
        __syncthreads();

        FragB16 a0, a1;
#pragma unroll
        for (int j = 0; j < 8; ++j) {
            int k = kAmap(j, hi8);
            a0.u[j] = *(const unsigned int*)&sA[lane16][k];
            a1.u[j] = *(const unsigned int*)&sA[lane16 + 16][k];
        }

        // B fragment directly from W rows (W is [N,K] row-major -> Bᵀ source)
        int n  = n0 + wave*16 + lane16;
        int kb = k0 + ((lane < 16) ? 0 : 16);
        FragB16 b = load_b_from_f32(W + (size_t)n*DIMV + kb);

        c0 = __builtin_amdgcn_wmma_f32_16x16x32_bf16(false, a0.v, false, b.v, (short)0, c0, false, false);
        c1 = __builtin_amdgcn_wmma_f32_16x16x32_bf16(false, a1.v, false, b.v, (short)0, c1, false, false);
        __syncthreads();
    }

    // bias (per output column)
    {
        float bv = bias[n0 + wave*16 + lane16];
#pragma unroll
        for (int r = 0; r < 8; ++r) { c0[r] += bv; c1[r] += bv; }
    }

    // spill C tile for cross-column RMS statistics
    {
        int col = wave*16 + lane16;
#pragma unroll
        for (int r = 0; r < 8; ++r) {
            sC[r + hi8][col]      = c0[r];
            sC[16 + r + hi8][col] = c1[r];
        }
    }
    __syncthreads();

    const int t = n0 >> 10;               // 0=q, 1=k, 2=v
    const int h = (n0 >> 6) & (NHEADS - 1);

    if (tid < 32) {
        float ss = 0.f;
#pragma unroll 8
        for (int c = 0; c < HDIM; ++c) { float v = sC[tid][c]; ss += v * v; }
        sRstd[tid] = (t == 2) ? 1.f : rsqrtf(ss * (1.f / HDIM) + EPSV);
    }
    __syncthreads();

    {
        int row = tid >> 2;
        int cb  = (tid & 3) * 16;
        int gm  = m0 + row;
        int bI  = gm / SS, sI = gm % SS;
        if (t == 2) {
            // transposed store: Vt[((b*H+h)*HDIM + dh)*S + s]
            size_t vbase = ((size_t)bI*NHEADS + h) * HDIM;
#pragma unroll
            for (int c = 0; c < 16; ++c) {
                int dh = cb + c;
                Vt[(vbase + dh)*SS + sI] = f2bf(sC[row][dh]);
            }
        } else {
            unsigned short* dst = (t == 0) ? Qn : Kn;
            const float smul = (t == 0) ? 0.125f : 1.f;   // Dh^-0.5 folded into q
            float rstd = sRstd[row];
            size_t base = (((size_t)bI*NHEADS + h)*SS + sI) * HDIM;
#pragma unroll
            for (int c = 0; c < 16; ++c) {
                int dh = cb + c;
                float wgt = (t == 0) ? qw[dh] : kw[dh];
                dst[base + dh] = f2bf(sC[row][dh] * rstd * wgt * smul);
            }
        }
    }
}

// ---------------------------------------------------------------------------
// Kernel 2: flash attention. One wave per 16-query tile; key chunks of 32.
// 4 score WMMAs + 4 PV WMMAs per chunk; online softmax in registers with
// __shfl_xor row reductions (rows live across 16-lane halves in C layout).
// K fragments and (transposed) V fragments are contiguous b128 loads.
// ---------------------------------------------------------------------------
__global__ __launch_bounds__(128)
void attn_kernel(const unsigned short* __restrict__ Qn,
                 const unsigned short* __restrict__ Kn,
                 const unsigned short* __restrict__ Vt,   // [B,H,HDIM,S]
                 const unsigned char*  __restrict__ mask, // [B,S] bool
                 unsigned short* __restrict__ O)          // [B,S,DIMV] bf16
{
    __shared__ unsigned short sP[4][16][32];  // per-wave P relayout buffer

    const int tid    = threadIdx.x;
    const int wave   = tid >> 5;
    const int lane   = tid & 31;
    const int lane16 = lane & 15;
    const int hi8    = (lane >> 4) << 3;
    const int h      = blockIdx.y;
    const int bI     = blockIdx.z;
    const int q0     = blockIdx.x * 64 + wave * 16;

    const size_t headBase = ((size_t)bI*NHEADS + h) * (size_t)SS * HDIM;

    // Q fragments (A 16x64 = two 16x32 halves over dh)
    FragB16 aq0, aq1;
    {
        const unsigned short* qrow = Qn + headBase + (size_t)(q0 + lane16)*HDIM;
#pragma unroll
        for (int j = 0; j < 8; ++j) {
            int k = kAmap(j, hi8);
            aq0.u[j] = *(const unsigned int*)&qrow[k];
            aq1.u[j] = *(const unsigned int*)&qrow[32 + k];
        }
    }

    float mrow[8], lrow[8];
    v8f oa[4] = {};
#pragma unroll
    for (int r = 0; r < 8; ++r) { mrow[r] = -3.0e38f; lrow[r] = 0.f; }

    const int kbL = (lane < 16) ? 0 : 16;

    for (int kc = 0; kc < SS; kc += 32) {
        // ---- scores: two 16x16 tiles over key columns kc..+15, kc+16..+31
        v8f cs[2];
#pragma unroll
        for (int g = 0; g < 2; ++g) {
            int key = kc + g*16 + lane16;
            const unsigned short* krow = Kn + headBase + (size_t)key*HDIM;
            FragB16 bk0 = load_b_contig_bf16(krow + kbL);
            FragB16 bk1 = load_b_contig_bf16(krow + 32 + kbL);
            v8f z = {};
            cs[g] = __builtin_amdgcn_wmma_f32_16x16x32_bf16(false, aq0.v, false, bk0.v, (short)0, z,     false, false);
            cs[g] = __builtin_amdgcn_wmma_f32_16x16x32_bf16(false, aq1.v, false, bk1.v, (short)0, cs[g], false, false);
            if (!mask[(size_t)bI*SS + key]) {
#pragma unroll
                for (int r = 0; r < 8; ++r) cs[g][r] = -3.0e38f;
            }
        }

        // ---- online softmax (row stats across 16-lane halves)
        float alpha[8];
#pragma unroll
        for (int r = 0; r < 8; ++r) {
            float v = fmaxf(cs[0][r], cs[1][r]);
            v = fmaxf(v, __shfl_xor(v, 1));
            v = fmaxf(v, __shfl_xor(v, 2));
            v = fmaxf(v, __shfl_xor(v, 4));
            v = fmaxf(v, __shfl_xor(v, 8));
            float mnew = fmaxf(mrow[r], v);
            alpha[r] = __expf(mrow[r] - mnew);
            float p0 = __expf(cs[0][r] - mnew);
            float p1 = __expf(cs[1][r] - mnew);
            cs[0][r] = p0; cs[1][r] = p1;
            float s = p0 + p1;
            s += __shfl_xor(s, 1);
            s += __shfl_xor(s, 2);
            s += __shfl_xor(s, 4);
            s += __shfl_xor(s, 8);
            lrow[r] = lrow[r]*alpha[r] + s;
            mrow[r] = mnew;
        }
#pragma unroll
        for (int n = 0; n < 4; ++n)
#pragma unroll
            for (int r = 0; r < 8; ++r) oa[n][r] *= alpha[r];

        // ---- relayout P (C layout -> A layout) through per-wave LDS
#pragma unroll
        for (int r = 0; r < 8; ++r) {
            sP[wave][r + hi8][lane16]      = f2bf(cs[0][r]);
            sP[wave][r + hi8][16 + lane16] = f2bf(cs[1][r]);
        }
        // same-wave DS ordering: LDS is in-order per wave; pin compiler ordering
        asm volatile("s_wait_dscnt 0" ::: "memory");

        FragB16 ap;
#pragma unroll
        for (int j = 0; j < 8; ++j) {
            int k = kAmap(j, hi8);
            ap.u[j] = *(const unsigned int*)&sP[wave][lane16][k];
        }
        asm volatile("" ::: "memory");   // don't sink next iteration's stores above loads

        // ---- O += P x V  (4 dh tiles); Vt rows are contiguous along keys
#pragma unroll
        for (int n = 0; n < 4; ++n) {
            int dh = n*16 + lane16;
            const unsigned short* vrow =
                Vt + (((size_t)bI*NHEADS + h)*HDIM + dh)*SS + kc + kbL;
            FragB16 bv = load_b_contig_bf16(vrow);
            oa[n] = __builtin_amdgcn_wmma_f32_16x16x32_bf16(false, ap.v, false, bv.v, (short)0, oa[n], false, false);
        }
    }

    // ---- normalize and store O as bf16 in [b][s][D] (proj GEMM A layout)
#pragma unroll
    for (int r = 0; r < 8; ++r) {
        float inv = (lrow[r] > 0.f) ? (1.f / lrow[r]) : 0.f;
        int gq = q0 + r + hi8;
        size_t base = ((size_t)bI*SS + gq)*DIMV + (size_t)h*HDIM;
#pragma unroll
        for (int n = 0; n < 4; ++n)
            O[base + n*16 + lane16] = f2bf(oa[n][r] * inv);
    }
}

// ---------------------------------------------------------------------------
// Kernel 3: output projection GEMM (bf16 A from attention, f32 W -> bf16 B,
// f32 accumulate) + bias, f32 output.
// ---------------------------------------------------------------------------
__global__ __launch_bounds__(128)
void proj_kernel(const unsigned short* __restrict__ A,   // [MTOT, DIMV] bf16
                 const float* __restrict__ W,            // [DIMV, DIMV] (e-major)
                 const float* __restrict__ bias,         // [DIMV]
                 float* __restrict__ Out)                 // [MTOT, DIMV]
{
    __shared__ unsigned short sA[32][32];

    const int tid    = threadIdx.x;
    const int wave   = tid >> 5;
    const int lane   = tid & 31;
    const int lane16 = lane & 15;
    const int hi8    = (lane >> 4) << 3;
    const int n0     = blockIdx.x * 64;
    const int m0     = blockIdx.y * 32;

    v8f c0 = {}; v8f c1 = {};

    for (int k0 = 0; k0 < DIMV; k0 += 32) {
        {
            int row = tid >> 2;
            int cb  = (tid & 3) * 8;
            uint4 v = *(const uint4*)(A + (size_t)(m0 + row)*DIMV + k0 + cb);
            *(uint4*)&sA[row][cb] = v;
            if (k0 + 32 < DIMV)
                __builtin_prefetch(A + (size_t)(m0 + row)*DIMV + k0 + 32 + cb, 0, 0);
        }
        __syncthreads();

        FragB16 a0, a1;
#pragma unroll
        for (int j = 0; j < 8; ++j) {
            int k = kAmap(j, hi8);
            a0.u[j] = *(const unsigned int*)&sA[lane16][k];
            a1.u[j] = *(const unsigned int*)&sA[lane16 + 16][k];
        }

        int n  = n0 + wave*16 + lane16;
        int kb = k0 + ((lane < 16) ? 0 : 16);
        FragB16 b = load_b_from_f32(W + (size_t)n*DIMV + kb);

        c0 = __builtin_amdgcn_wmma_f32_16x16x32_bf16(false, a0.v, false, b.v, (short)0, c0, false, false);
        c1 = __builtin_amdgcn_wmma_f32_16x16x32_bf16(false, a1.v, false, b.v, (short)0, c1, false, false);
        __syncthreads();
    }

    int gn = n0 + wave*16 + lane16;
    float bv = bias[gn];
#pragma unroll
    for (int r = 0; r < 8; ++r) {
        Out[(size_t)(m0 + r + hi8)*DIMV + gn]      = c0[r] + bv;
        Out[(size_t)(m0 + 16 + r + hi8)*DIMV + gn] = c1[r] + bv;
    }
}

// ---------------------------------------------------------------------------
extern "C" void kernel_launch(void* const* d_in, const int* in_sizes, int n_in,
                              void* d_out, int out_size, void* d_ws, size_t ws_size,
                              hipStream_t stream) {
    (void)in_sizes; (void)n_in; (void)out_size; (void)ws_size;
    const float*         x      = (const float*)d_in[0];
    const unsigned char* amask  = (const unsigned char*)d_in[1];
    const float*         w_qkv  = (const float*)d_in[2];
    const float*         b_qkv  = (const float*)d_in[3];
    const float*         qw     = (const float*)d_in[4];
    const float*         kw     = (const float*)d_in[5];
    const float*         w_proj = (const float*)d_in[6];
    const float*         b_proj = (const float*)d_in[7];
    float*               out    = (float*)d_out;

    // workspace: 4 x 16MB bf16 staging buffers (Q̂, K̂, Vᵀ, attn-out) = 64MB
    unsigned short* ws  = (unsigned short*)d_ws;
    const size_t per    = (size_t)BB * NHEADS * SS * HDIM;  // 8,388,608 elems
    unsigned short* Qn  = ws;
    unsigned short* Kn  = Qn + per;
    unsigned short* Vt  = Kn + per;
    unsigned short* Oat = Vt + per;

    qkv_norm_kernel<<<dim3(3*DIMV/64, MTOT/32), 128, 0, stream>>>(
        x, w_qkv, b_qkv, qw, kw, Qn, Kn, Vt);
    attn_kernel<<<dim3(SS/64, NHEADS, BB), 128, 0, stream>>>(
        Qn, Kn, Vt, amask, Oat);
    proj_kernel<<<dim3(DIMV/64, MTOT/32), 128, 0, stream>>>(
        Oat, w_proj, b_proj, out);
}
